// GraphAttention_25735444037843
// MI455X (gfx1250) — matrix operand order repaired and checked by reading, not compiled, not run on hardware
//
#include <hip/hip_runtime.h>

typedef __attribute__((ext_vector_type(16))) _Float16 v16h;
typedef __attribute__((ext_vector_type(8)))  _Float16 v8h;
typedef __attribute__((ext_vector_type(8)))  float    v8f;
typedef __attribute__((ext_vector_type(2)))  float    v2f;

#define Nn 4096
#define Ff 128
#define Fo 64
#define Hh 8
#define LOG2E 1.44269504088896340736f
#define NEG_BIG_C (-1e10f)
#define SHIFT_C 10.0f
#define LEAKY_A 0.2f

// ---------------------------------------------------------------------------
// Phase 1 (fp32 WMMA): feat tile [16 nodes x 64 outs] = X[16x128] @ W[h][128x64]
// via 32 chained V_WMMA_F32_16X16X4_F32 (full fp32 precision).
// Emits featT16[h][o][n] (K-contiguous f16 B-operand storage for phase 2) and
// s[h][n] = feat.a_self, t[h][n] = feat.a_neigh via in-wave shuffle reduction.
// One wave per (head, 16-node tile); 4 waves/block.
// ---------------------------------------------------------------------------
__global__ __launch_bounds__(128)
void gat_proj_wmma_kernel(const float* __restrict__ X,
                          const float* __restrict__ W,
                          const float* __restrict__ a_self,
                          const float* __restrict__ a_neigh,
                          _Float16* __restrict__ featT,
                          float* __restrict__ s,
                          float* __restrict__ t)
{
    int wave = blockIdx.x * (blockDim.x >> 5) + (threadIdx.x >> 5);
    int h    = wave >> 8;            // 256 node-tiles per head
    int nt   = wave & 255;
    int n0   = nt * 16;
    int lane = threadIdx.x & 31;
    int half = lane >> 4;            // K-half ownership (A/B layout)
    int c    = lane & 15;            // A row (M) / B,C,D column

    const float* xrow = X + (size_t)(n0 + c) * Ff;   // A operand: row M = c
    const float* Wh   = W + (size_t)h * Ff * Fo;

    v8f acc0 = {}, acc1 = {}, acc2 = {}, acc3 = {};

#pragma unroll 4
    for (int k0 = 0; k0 < Ff; k0 += 4) {
        int k = k0 + 2 * half;                       // this lane's two K's
        v2f a = *(const v2f*)(xrow + k);             // X[n0+c][k], X[n0+c][k+1]
        const float* wb = Wh + (size_t)k * Fo + c;   // W[h][k][col]
        v2f b0 = { wb[0],  wb[Fo +  0] };
        v2f b1 = { wb[16], wb[Fo + 16] };
        v2f b2 = { wb[32], wb[Fo + 32] };
        v2f b3 = { wb[48], wb[Fo + 48] };
        acc0 = __builtin_amdgcn_wmma_f32_16x16x4_f32(false, a, false, b0, (short)0, acc0, false, false);
        acc1 = __builtin_amdgcn_wmma_f32_16x16x4_f32(false, a, false, b1, (short)0, acc1, false, false);
        acc2 = __builtin_amdgcn_wmma_f32_16x16x4_f32(false, a, false, b2, (short)0, acc2, false, false);
        acc3 = __builtin_amdgcn_wmma_f32_16x16x4_f32(false, a, false, b3, (short)0, acc3, false, false);
    }

    // ---- featT f16 stores: lane owns col o = ot*16+c; its 8 VGPRs are nodes
    //      n0 + 8*half + (0..7), contiguous in featT[h][o][n] -> one b128 each.
    {
        v8f accs[4] = {acc0, acc1, acc2, acc3};
#pragma unroll
        for (int ot = 0; ot < 4; ++ot) {
            v8h hv;
#pragma unroll
            for (int e = 0; e < 8; ++e) hv[e] = (_Float16)accs[ot][e];
            *(v8h*)(featT + ((size_t)h * Fo + ot * 16 + c) * Nn + n0 + 8 * half) = hv;
        }
    }

    // ---- s,t: reduce over the 64 output features (4 o-tiles x 16 lanes).
    float as0 = a_self [(size_t)h * Fo +  0 + c];
    float as1 = a_self [(size_t)h * Fo + 16 + c];
    float as2 = a_self [(size_t)h * Fo + 32 + c];
    float as3 = a_self [(size_t)h * Fo + 48 + c];
    float an0 = a_neigh[(size_t)h * Fo +  0 + c];
    float an1 = a_neigh[(size_t)h * Fo + 16 + c];
    float an2 = a_neigh[(size_t)h * Fo + 32 + c];
    float an3 = a_neigh[(size_t)h * Fo + 48 + c];

    float sp[8], tp[8];
#pragma unroll
    for (int r = 0; r < 8; ++r) {
        sp[r] = fmaf(acc0[r], as0, fmaf(acc1[r], as1, fmaf(acc2[r], as2, acc3[r] * as3)));
        tp[r] = fmaf(acc0[r], an0, fmaf(acc1[r], an1, fmaf(acc2[r], an2, acc3[r] * an3)));
    }
#pragma unroll
    for (int m = 1; m <= 8; m <<= 1) {
#pragma unroll
        for (int r = 0; r < 8; ++r) {
            sp[r] += __shfl_xor(sp[r], m);
            tp[r] += __shfl_xor(tp[r], m);
        }
    }
    if (c == 0) {                    // rows 8*half + (0..7)
        float* so = s + (size_t)h * Nn + n0 + 8 * half;
        float* to = t + (size_t)h * Nn + n0 + 8 * half;
#pragma unroll
        for (int r = 0; r < 8; ++r) { so[r] = sp[r]; to[r] = tp[r]; }
    }
}

// ---------------------------------------------------------------------------
// Phase 2: fused masked-softmax attention + attn@feat via WMMA f16->f32.
// One wave per (head, 16-row tile). Scores generated directly in the 16-bit
// A-matrix VGPR layout; featT supplies B operands; 4 accumulators cover the
// 16x64 output tile. Fixed-shift exp (no running max) is safe: scores ~
// N(0,2); masked entries -> exp(-1e10) == 0 exactly.
// ---------------------------------------------------------------------------
__device__ __forceinline__ void score4(float s_lane, float4 tv, float4 av,
                                       float* p, float& lsum)
{
    float tj[4] = {tv.x, tv.y, tv.z, tv.w};
    float aj[4] = {av.x, av.y, av.z, av.w};
#pragma unroll
    for (int e = 0; e < 4; ++e) {
        float sc = s_lane + tj[e];
        sc = (sc >= 0.f) ? sc : LEAKY_A * sc;           // leaky relu
        sc = fmaf(NEG_BIG_C, 1.0f - aj[e], sc);         // adjacency mask
        float pe = __builtin_amdgcn_exp2f((sc - SHIFT_C) * LOG2E);
        p[e] = pe;
        lsum += pe;
    }
}

__global__ __launch_bounds__(128)
void gat_attn_kernel(const float* __restrict__ A,
                     const float* __restrict__ s,
                     const float* __restrict__ t,
                     const _Float16* __restrict__ featT,
                     const float* __restrict__ b,
                     float* __restrict__ out)
{
    int wave = blockIdx.x * (blockDim.x >> 5) + (threadIdx.x >> 5);
    int h    = wave >> 8;            // 256 row-tiles per head
    int it   = wave & 255;
    int i0   = it * 16;
    int lane = threadIdx.x & 31;
    int half = lane >> 4;            // K-half ownership (A layout)
    int r    = lane & 15;            // row owned in score phase / col in C/D

    float s_lane = s[(size_t)h * Nn + i0 + r];
    const float*    arow = A + (size_t)(i0 + r) * Nn;
    const float*    th   = t + (size_t)h * Nn;
    const _Float16* fT   = featT + (size_t)h * Fo * Nn;

    v8f acc0 = {}, acc1 = {}, acc2 = {}, acc3 = {};
    float lsum = 0.f;

    for (int jt = 0; jt < Nn; jt += 32) {
        int jb0 = jt + half * 16;         // K in {16*half .. 16*half+7}
        int jb1 = jt + half * 16 + 8;     // K in {16*half+8 .. +7}

        float4 a0 = *(const float4*)(arow + jb0);
        float4 a1 = *(const float4*)(arow + jb0 + 4);
        float4 a2 = *(const float4*)(arow + jb1);
        float4 a3 = *(const float4*)(arow + jb1 + 4);
        float4 t0 = *(const float4*)(th + jb0);
        float4 t1 = *(const float4*)(th + jb0 + 4);
        float4 t2 = *(const float4*)(th + jb1);
        float4 t3 = *(const float4*)(th + jb1 + 4);

        float p[16];
        score4(s_lane, t0, a0, p + 0,  lsum);
        score4(s_lane, t1, a1, p + 4,  lsum);
        score4(s_lane, t2, a2, p + 8,  lsum);
        score4(s_lane, t3, a3, p + 12, lsum);

        v16h aop;
#pragma unroll
        for (int e = 0; e < 16; ++e) aop[e] = (_Float16)p[e];

        // B operands: lane -> col o = ot*16 + r, K = jt + 16*half + (0..15)
        const _Float16* bbase = fT + (size_t)r * Nn + jt + half * 16;
        v16h b0 = *(const v16h*)(bbase);
        v16h b1 = *(const v16h*)(bbase + (size_t)16 * Nn);
        v16h b2 = *(const v16h*)(bbase + (size_t)32 * Nn);
        v16h b3 = *(const v16h*)(bbase + (size_t)48 * Nn);

        acc0 = __builtin_amdgcn_wmma_f32_16x16x32_f16(false, aop, false, b0, (short)0, acc0, false, false);
        acc1 = __builtin_amdgcn_wmma_f32_16x16x32_f16(false, aop, false, b1, (short)0, acc1, false, false);
        acc2 = __builtin_amdgcn_wmma_f32_16x16x32_f16(false, aop, false, b2, (short)0, acc2, false, false);
        acc3 = __builtin_amdgcn_wmma_f32_16x16x32_f16(false, aop, false, b3, (short)0, acc3, false, false);
    }

    // complete per-row softmax denominators (row i split across lane & lane+16)
    lsum += __shfl_xor(lsum, 16);

    // C/D layout: VGPR r8 holds row M = r8 + 8*half; row-M denom lives in lane M
    float inv[8];
#pragma unroll
    for (int r8 = 0; r8 < 8; ++r8)
        inv[r8] = 1.0f / __shfl(lsum, r8 + 8 * half);

    float bv0 = b[(size_t)h * Fo +  0 + r];
    float bv1 = b[(size_t)h * Fo + 16 + r];
    float bv2 = b[(size_t)h * Fo + 32 + r];
    float bv3 = b[(size_t)h * Fo + 48 + r];

#pragma unroll
    for (int r8 = 0; r8 < 8; ++r8) {
        int m = r8 + 8 * half;
        float* op = out + (size_t)(i0 + m) * (Hh * Fo) + (size_t)h * Fo + r;
        float v0 = fmaf(acc0[r8], inv[r8], bv0); op[0]  = v0 > 0.f ? v0 : 0.f;
        float v1 = fmaf(acc1[r8], inv[r8], bv1); op[16] = v1 > 0.f ? v1 : 0.f;
        float v2 = fmaf(acc2[r8], inv[r8], bv2); op[32] = v2 > 0.f ? v2 : 0.f;
        float v3 = fmaf(acc3[r8], inv[r8], bv3); op[48] = v3 > 0.f ? v3 : 0.f;
    }
}

// ---------------------------------------------------------------------------
extern "C" void kernel_launch(void* const* d_in, const int* in_sizes, int n_in,
                              void* d_out, int out_size, void* d_ws, size_t ws_size,
                              hipStream_t stream)
{
    const float* X       = (const float*)d_in[0];   // [N,F]
    const float* A       = (const float*)d_in[1];   // [N,N]
    const float* W       = (const float*)d_in[2];   // [H,F,F_]
    const float* a_self  = (const float*)d_in[3];   // [H,F_]
    const float* a_neigh = (const float*)d_in[4];   // [H,F_]
    const float* b       = (const float*)d_in[5];   // [H,F_]
    float* out = (float*)d_out;                     // [N, H*F_]

    char* ws = (char*)d_ws;
    size_t featT_bytes = (size_t)Hh * Nn * Fo * sizeof(_Float16);  // 4 MB
    _Float16* featT = (_Float16*)ws;
    float*    s     = (float*)(ws + featT_bytes);
    float*    t     = s + (size_t)Hh * Nn;

    // 2048 waves each: phase 1 = one 16-node tile, phase 2 = one 16-row tile.
    gat_proj_wmma_kernel<<<(Hh * (Nn / 16)) / 4, 128, 0, stream>>>(
        X, W, a_self, a_neigh, featT, s, t);
    gat_attn_kernel<<<(Hh * (Nn / 16)) / 4, 128, 0, stream>>>(
        A, s, t, featT, b, out);
}
